// Model_84018150244606
// MI455X (gfx1250) — matrix operand order repaired and checked by reading, not compiled, not run on hardware
//
#include <hip/hip_runtime.h>
#include <hip/hip_bf16.h>

typedef __bf16 bf16;
typedef __attribute__((ext_vector_type(16))) __bf16 v16bf;
typedef __attribute__((ext_vector_type(8)))  float  v8f;

#define DEV __device__ __forceinline__

// -------- problem constants (from reference setup_inputs) --------
// x: [8,384,128,128]; stem -> [8,96,32,32]; N=1024, nH=4, hd=24, rh=32
constexpr int BN   = 8;
constexpr int DCH  = 384;
constexpr int HIMG = 128;
constexpr int C    = 96;
constexpr int HQ   = 32;
constexpr int NQ   = 1024;   // HQ*HQ
constexpr int NH   = 4;
constexpr int HD   = 24;
constexpr int HDP  = 32;     // padded head dim for WMMA K
constexpr int ROWS = BN * NQ; // 8192 token rows

DEV v8f zero8() {
  v8f z;
  #pragma unroll
  for (int i = 0; i < 8; ++i) z[i] = 0.f;
  return z;
}

DEV v8f wmma_bf16(v16bf a, v16bf b, v8f c) {
  // (neg_a, A, neg_b, B, c_mod, C, reuse_a, reuse_b)
  return __builtin_amdgcn_wmma_f32_16x16x32_bf16(false, a, false, b, (short)0, c,
                                                 false, false);
}

// A fragment: row-major A[16][K], row stride ldk, tile at `base`.
// Lane l holds row (l&15); K = (l<16 ? 0..15 : 16..31).
DEV v16bf load_frag_a(const bf16* base, int ldk, int lane) {
  v16bf f;
  int m = lane & 15;
  int koff = (lane < 16) ? 0 : 16;
  #pragma unroll
  for (int r = 0; r < 8; ++r) {
    int kb = 2 * r + koff;
    f[2 * r]     = base[(size_t)m * ldk + kb];
    f[2 * r + 1] = base[(size_t)m * ldk + kb + 1];
  }
  return f;
}

// B fragment with K contiguous in memory: element(k,n) = base[n*ldcol + k].
// Lane l holds column (l&15); K = (l<16 ? 0..15 : 16..31). One 32B vector load.
DEV v16bf load_frag_b_kmajor(const bf16* base, int ldcol, int lane) {
  int n = lane & 15;
  int kb = (lane < 16) ? 0 : 16;
  return *reinterpret_cast<const v16bf*>(base + (size_t)n * ldcol + kb);
}

DEV float bn_relu6(float v, const float* __restrict__ bn, int c) {
  float g = bn[c], be = bn[C + c], mn = bn[2 * C + c], vr = bn[3 * C + c];
  float r = (v - mn) * (g * rsqrtf(vr + 1e-5f)) + be;
  return fminf(fmaxf(r, 0.f), 6.f);
}

// -------- weight conversion (pre-transposed for K-contiguous B fragments) ----
__global__ void k_cvt(const float* __restrict__ wq, const float* __restrict__ wkv,
                      const float* __restrict__ wproj, const float* __restrict__ wout,
                      bf16* __restrict__ wqT, bf16* __restrict__ wkvT,
                      bf16* __restrict__ wprojT, bf16* __restrict__ woutB) {
  int t = blockIdx.x * 256 + threadIdx.x;
  if (t < C * C) {           // [96,96] -> [n][k]
    int k = t / C, n = t % C;
    wqT[n * C + k]    = (bf16)wq[k * C + n];
    wprojT[n * C + k] = (bf16)wproj[k * C + n];
  }
  if (t < C * 2 * C) {       // [96,192] -> [n(192)][k(96)]
    int k = t / (2 * C), n = t % (2 * C);
    wkvT[n * C + k] = (bf16)wkv[k * 2 * C + n];
  }
  if (t < DCH * C) {         // w_out already [oc][c] == [n][k]
    woutB[t] = (bf16)wout[t];
  }
}

// -------- relative-position bias materialization: biasT[h][n][m] -------------
__global__ void k_bias(const float* __restrict__ table, float* __restrict__ biasT) {
  size_t tid = (size_t)blockIdx.x * 256 + threadIdx.x;
  if (tid >= (size_t)NH * NQ * NQ) return;
  int m = tid & 1023;
  int n = (tid >> 10) & 1023;
  int h = tid >> 20;
  int i = n >> 5, j = n & 31, p = m >> 5, q = m & 31;
  int idx = (i - p + 31) * 63 + (j - q + 31);
  biasT[tid] = table[idx * NH + h];
}

// -------- fused stem: read x once; conv4x4/s4 + maxpool/avg + grouped 1x1 ----
__global__ void k_stem(const float* __restrict__ x,
                       const float* __restrict__ w_le, const float* __restrict__ b_le,
                       const float* __restrict__ bn_le,
                       const float* __restrict__ w_max, const float* __restrict__ b_max,
                       const float* __restrict__ bn_max,
                       const float* __restrict__ w_avg, const float* __restrict__ b_avg,
                       const float* __restrict__ bn_avg,
                       bf16* __restrict__ xq, bf16* __restrict__ tmx,
                       bf16* __restrict__ tav) {
  int tid = blockIdx.x * 256 + threadIdx.x;
  if (tid >= BN * C * HQ * HQ) return;
  int ow = tid & 31;
  int oh = (tid >> 5) & 31;
  int c  = (tid >> 10) % C;
  int b  = tid / (NQ * C);

  float conv = 0.f, mg[4], ag[4];
  #pragma unroll
  for (int i = 0; i < 4; ++i) {
    int ch = 4 * c + i;
    float mx = -3.4e38f, sm = 0.f;
    #pragma unroll
    for (int ky = 0; ky < 4; ++ky) {
      float4 xv = *reinterpret_cast<const float4*>(
          x + (((size_t)(b * DCH + ch) * HIMG + (4 * oh + ky)) * HIMG + 4 * ow));
      float xs[4] = {xv.x, xv.y, xv.z, xv.w};
      #pragma unroll
      for (int kx = 0; kx < 4; ++kx) {
        float v = xs[kx];
        conv += w_le[((c * 4 + i) * 4 + ky) * 4 + kx] * v;
        mx = fmaxf(mx, v);
        sm += v;
      }
    }
    mg[i] = mx;
    ag[i] = sm * (1.f / 16.f);
  }
  float xqv = bn_relu6(conv + b_le[c], bn_le, c);
  float mv = 0.f, av = 0.f;
  #pragma unroll
  for (int i = 0; i < 4; ++i) {
    mv += w_max[c * 4 + i] * mg[i];
    av += w_avg[c * 4 + i] * ag[i];
  }
  mv = bn_relu6(mv + b_max[c], bn_max, c);
  av = bn_relu6(av + b_avg[c], bn_avg, c);

  size_t row = (size_t)b * NQ + oh * HQ + ow;
  xq[row * C + c]  = (bf16)xqv;
  tmx[row * C + c] = (bf16)mv;
  tav[row * C + c] = (bf16)av;
}

// -------- q / k / v projections (WMMA bf16) ---------------------------------
// q:  [8192,96]@[96,96]  -> q[b][h][n][32] (scaled by hd^-0.5, padded K)
// kv: [8192,96]@[96,192] -> k[b][h][n][32] ; v stored transposed [b][h][32][1024]
__global__ void __launch_bounds__(32) k_qkv(
    const bf16* __restrict__ xq, const bf16* __restrict__ tmx, const bf16* __restrict__ tav,
    const bf16* __restrict__ wqT, const bf16* __restrict__ wkvT,
    bf16* __restrict__ qm, bf16* __restrict__ k1, bf16* __restrict__ v1,
    bf16* __restrict__ k2, bf16* __restrict__ v2) {
  const int lane = threadIdx.x;
  const int row0 = blockIdx.x * 16;
  const int b = row0 >> 10;
  const float qscale = rsqrtf((float)HD);

  // ---- q ----
  for (int nt = 0; nt < 6; ++nt) {
    v8f acc = zero8();
    #pragma unroll
    for (int ks = 0; ks < 3; ++ks) {
      v16bf a  = load_frag_a(xq + (size_t)row0 * C + ks * 32, C, lane);
      v16bf bm = load_frag_b_kmajor(wqT + (size_t)(nt * 16) * C + ks * 32, C, lane);
      acc = wmma_bf16(a, bm, acc);
    }
    int col = nt * 16 + (lane & 15);
    int h = col / HD, d = col % HD;
    #pragma unroll
    for (int r = 0; r < 8; ++r) {
      int m = r + ((lane < 16) ? 0 : 8);
      int n = (row0 + m) & 1023;
      qm[(((size_t)(b * NH + h) * NQ + n) << 5) + d] = (bf16)(acc[r] * qscale);
    }
  }
  // ---- kv (two branches) ----
  for (int br = 0; br < 2; ++br) {
    const bf16* src = br ? tav : tmx;
    bf16* kd = br ? k2 : k1;
    bf16* vd = br ? v2 : v1;
    for (int nt = 0; nt < 12; ++nt) {
      v8f acc = zero8();
      #pragma unroll
      for (int ks = 0; ks < 3; ++ks) {
        v16bf a  = load_frag_a(src + (size_t)row0 * C + ks * 32, C, lane);
        v16bf bm = load_frag_b_kmajor(wkvT + (size_t)(nt * 16) * C + ks * 32, C, lane);
        acc = wmma_bf16(a, bm, acc);
      }
      int j = nt * 16 + (lane & 15);
      int kv = j / C, jj = j % C, h = jj / HD, d = jj % HD;
      #pragma unroll
      for (int r = 0; r < 8; ++r) {
        int m = r + ((lane < 16) ? 0 : 8);
        int n = (row0 + m) & 1023;
        if (kv == 0)
          kd[(((size_t)(b * NH + h) * NQ + n) << 5) + d] = (bf16)acc[r];
        else
          vd[(((size_t)(b * NH + h) * HDP + d) << 10) + n] = (bf16)acc[r];
      }
    }
  }
}

// -------- attention: Q*K^T + bias -> softmax -> P*V (flash-style, LDS row) ---
__global__ void __launch_bounds__(32) k_attn(
    const bf16* __restrict__ qm, const bf16* __restrict__ k1, const bf16* __restrict__ v1,
    const bf16* __restrict__ k2, const bf16* __restrict__ v2,
    const float* __restrict__ biasT, float* __restrict__ o1, float* __restrict__ o2) {
  __shared__ float sP[16 * NQ];  // 64 KB: 16 query rows x 1024 keys
  const int lane = threadIdx.x;
  int bid = blockIdx.x;
  int mt = bid & 63;
  int bh = (bid >> 6) & 31;
  int br = bid >> 11;
  int b = bh >> 2, h = bh & 3;
  const bf16* kp = br ? k2 : k1;
  const bf16* vp = br ? v2 : v1;
  float* op = br ? o2 : o1;

  v16bf aq = load_frag_a(qm + (((size_t)(b * NH + h) * NQ + mt * 16) << 5), HDP, lane);
  const bf16*  kbase = kp + (((size_t)(b * NH + h) * NQ) << 5);
  const float* bbase = biasT + ((size_t)h * NQ + mt * 16) * NQ;

  // logits
  for (int nt = 0; nt < 64; ++nt) {
    v16bf bk = load_frag_b_kmajor(kbase + (size_t)(nt * 16) * HDP, HDP, lane);
    v8f c = zero8();
    c = wmma_bf16(aq, bk, c);
    int n = nt * 16 + (lane & 15);
    #pragma unroll
    for (int r = 0; r < 8; ++r) {
      int lm = r + ((lane < 16) ? 0 : 8);
      sP[lm * NQ + n] = c[r] + bbase[(size_t)lm * NQ + n];
    }
  }
  __syncthreads();

  // softmax: lane handles row (lane&15), half (lane>>4); combine via shfl_xor(16)
  int rm = lane & 15, half = lane >> 4;
  float mx = -3.4e38f;
  for (int j = 0; j < 512; ++j) mx = fmaxf(mx, sP[rm * NQ + half * 512 + j]);
  mx = fmaxf(mx, __shfl_xor(mx, 16));
  float s = 0.f;
  for (int j = 0; j < 512; ++j) {
    int idx = rm * NQ + half * 512 + j;
    float e = __expf(sP[idx] - mx);
    sP[idx] = e;
    s += e;
  }
  s += __shfl_xor(s, 16);
  float rinv = 1.f / s;
  __syncthreads();

  // P @ V  (V transposed: [32][1024], K-contiguous B fragments)
  v8f acc0 = zero8(), acc1 = zero8();
  const bf16* vbase = vp + (((size_t)(b * NH + h) * HDP) << 10);
  for (int kb = 0; kb < 32; ++kb) {
    v16bf ap;
    {
      int m = lane & 15;
      int koff = (lane < 16) ? 0 : 16;
      #pragma unroll
      for (int r = 0; r < 8; ++r) {
        int kk = 2 * r + koff;
        ap[2 * r]     = (bf16)(sP[m * NQ + kb * 32 + kk] * rinv);
        ap[2 * r + 1] = (bf16)(sP[m * NQ + kb * 32 + kk + 1] * rinv);
      }
    }
    v16bf b0 = load_frag_b_kmajor(vbase + kb * 32, NQ, lane);
    v16bf b1 = load_frag_b_kmajor(vbase + ((size_t)16 << 10) + kb * 32, NQ, lane);
    acc0 = wmma_bf16(ap, b0, acc0);
    acc1 = wmma_bf16(ap, b1, acc1);
  }

  int dcol = lane & 15;
  #pragma unroll
  for (int r = 0; r < 8; ++r) {
    int m = r + ((lane < 16) ? 0 : 8);
    size_t row = (size_t)b * NQ + mt * 16 + m;
    op[row * C + h * HD + dcol] = acc0[r];
    if (dcol < 8) op[row * C + h * HD + 16 + dcol] = acc1[r];
  }
}

// -------- (o1+o2)@w_proj (+2*b_proj) -> @w_out^T (+b_out) -> y[b][384][1024] -
__global__ void __launch_bounds__(32) k_proj_out(
    const float* __restrict__ o1, const float* __restrict__ o2,
    const bf16* __restrict__ wprojT, const float* __restrict__ b_proj,
    const bf16* __restrict__ woutB, const float* __restrict__ b_out,
    float* __restrict__ y) {
  __shared__ bf16 sT[16 * C];
  const int lane = threadIdx.x;
  const int row0 = blockIdx.x * 16;
  const int b = row0 >> 10;

  for (int nt = 0; nt < 6; ++nt) {
    v8f acc = zero8();
    #pragma unroll
    for (int ks = 0; ks < 3; ++ks) {
      v16bf a;
      {
        int m = lane & 15;
        int koff = (lane < 16) ? 0 : 16;
        size_t base = (size_t)(row0 + m) * C + ks * 32;
        #pragma unroll
        for (int r = 0; r < 8; ++r) {
          int kk = 2 * r + koff;
          a[2 * r]     = (bf16)(o1[base + kk] + o2[base + kk]);
          a[2 * r + 1] = (bf16)(o1[base + kk + 1] + o2[base + kk + 1]);
        }
      }
      v16bf bm = load_frag_b_kmajor(wprojT + (size_t)(nt * 16) * C + ks * 32, C, lane);
      acc = wmma_bf16(a, bm, acc);
    }
    int col = nt * 16 + (lane & 15);
    float bp = 2.f * b_proj[col];
    #pragma unroll
    for (int r = 0; r < 8; ++r) {
      int lm = r + ((lane < 16) ? 0 : 8);
      sT[lm * C + col] = (bf16)(acc[r] + bp);
    }
  }
  __syncthreads();

  for (int nt = 0; nt < 24; ++nt) {
    v8f acc = zero8();
    #pragma unroll
    for (int ks = 0; ks < 3; ++ks) {
      v16bf a  = load_frag_a(sT + ks * 32, C, lane);
      v16bf bm = load_frag_b_kmajor(woutB + (size_t)(nt * 16) * C + ks * 32, C, lane);
      acc = wmma_bf16(a, bm, acc);
    }
    int oc = nt * 16 + (lane & 15);
    float bo = b_out[oc];
    #pragma unroll
    for (int r = 0; r < 8; ++r) {
      int m = r + ((lane < 16) ? 0 : 8);
      int sp = (row0 + m) & 1023;
      y[((size_t)(b * DCH + oc) << 10) + sp] = acc[r] + bo;
    }
  }
}

// -------- x4 bilinear upsample, align_corners=True ---------------------------
__global__ void k_upsample(const float* __restrict__ y, float* __restrict__ out) {
  size_t tid = (size_t)blockIdx.x * 256 + threadIdx.x;
  if (tid >= (size_t)BN * DCH * HIMG * HIMG) return;
  int ow = tid & 127;
  int oh = (tid >> 7) & 127;
  int ch = (int)((tid >> 14) % DCH);
  int b  = (int)(tid / ((size_t)DCH * HIMG * HIMG));
  const float sc = 31.f / 127.f;
  float sy = oh * sc, sx = ow * sc;
  int y0 = (int)sy, x0 = (int)sx;
  int y1 = min(y0 + 1, 31), x1 = min(x0 + 1, 31);
  float wy = sy - y0, wx = sx - x0;
  const float* p = y + ((size_t)(b * DCH + ch) << 10);
  float v00 = p[y0 * 32 + x0], v01 = p[y0 * 32 + x1];
  float v10 = p[y1 * 32 + x0], v11 = p[y1 * 32 + x1];
  out[tid] = (1.f - wy) * ((1.f - wx) * v00 + wx * v01) +
             wy * ((1.f - wx) * v10 + wx * v11);
}

extern "C" void kernel_launch(void* const* d_in, const int* in_sizes, int n_in,
                              void* d_out, int out_size, void* d_ws, size_t ws_size,
                              hipStream_t stream) {
  (void)in_sizes; (void)n_in; (void)out_size; (void)ws_size;
  const float* x      = (const float*)d_in[0];
  const float* w_le   = (const float*)d_in[1];
  const float* b_le   = (const float*)d_in[2];
  const float* bn_le  = (const float*)d_in[3];
  const float* w_max  = (const float*)d_in[4];
  const float* b_max  = (const float*)d_in[5];
  const float* bn_max = (const float*)d_in[6];
  const float* w_avg  = (const float*)d_in[7];
  const float* b_avg  = (const float*)d_in[8];
  const float* bn_avg = (const float*)d_in[9];
  const float* table  = (const float*)d_in[10];
  const float* w_q    = (const float*)d_in[11];
  const float* w_kv   = (const float*)d_in[12];
  const float* w_proj = (const float*)d_in[13];
  const float* b_proj = (const float*)d_in[14];
  const float* w_out  = (const float*)d_in[15];
  const float* b_out  = (const float*)d_in[16];
  float* out = (float*)d_out;

  char* ws = (char*)d_ws;
  size_t off = 0;
  auto take = [&](size_t bytes) -> char* {
    char* p = ws + off;
    off = (off + bytes + 255) & ~(size_t)255;
    return p;
  };
  float* biasT  = (float*)take((size_t)NH * NQ * NQ * 4);            // 16 MB
  bf16* xq      = (bf16*)take((size_t)ROWS * C * 2);
  bf16* tmx     = (bf16*)take((size_t)ROWS * C * 2);
  bf16* tav     = (bf16*)take((size_t)ROWS * C * 2);
  bf16* wqT     = (bf16*)take((size_t)C * C * 2);
  bf16* wkvT    = (bf16*)take((size_t)2 * C * C * 2);
  bf16* wprojT  = (bf16*)take((size_t)C * C * 2);
  bf16* woutB   = (bf16*)take((size_t)DCH * C * 2);
  size_t qkvB   = (size_t)BN * NH * NQ * HDP * 2;                    // 2 MB each
  bf16* qm = (bf16*)take(qkvB);
  bf16* k1 = (bf16*)take(qkvB);
  bf16* v1 = (bf16*)take(qkvB);
  bf16* k2 = (bf16*)take(qkvB);
  bf16* v2 = (bf16*)take(qkvB);
  float* o1 = (float*)take((size_t)ROWS * C * 4);
  float* o2 = (float*)take((size_t)ROWS * C * 4);
  float* yf = (float*)take((size_t)BN * DCH * NQ * 4);               // 12.6 MB

  k_cvt<<<(DCH * C + 255) / 256, 256, 0, stream>>>(w_q, w_kv, w_proj, w_out,
                                                   wqT, wkvT, wprojT, woutB);
  k_bias<<<(NH * NQ * NQ) / 256, 256, 0, stream>>>(table, biasT);
  k_stem<<<(BN * C * NQ) / 256, 256, 0, stream>>>(x, w_le, b_le, bn_le,
                                                  w_max, b_max, bn_max,
                                                  w_avg, b_avg, bn_avg,
                                                  xq, tmx, tav);
  // zero padded head-dim tails (K padding must be exact zeros for Q*K^T)
  (void)hipMemsetAsync(qm, 0, qkvB, stream);
  (void)hipMemsetAsync(k1, 0, qkvB, stream);
  (void)hipMemsetAsync(v1, 0, qkvB, stream);
  (void)hipMemsetAsync(k2, 0, qkvB, stream);
  (void)hipMemsetAsync(v2, 0, qkvB, stream);

  k_qkv<<<ROWS / 16, 32, 0, stream>>>(xq, tmx, tav, wqT, wkvT, qm, k1, v1, k2, v2);
  k_attn<<<2 * BN * NH * (NQ / 16), 32, 0, stream>>>(qm, k1, v1, k2, v2, biasT, o1, o2);
  k_proj_out<<<ROWS / 16, 32, 0, stream>>>(o1, o2, wprojT, b_proj, woutB, b_out, yf);
  k_upsample<<<(BN * DCH * HIMG * HIMG) / 256, 256, 0, stream>>>(yf, out);
}
